// PWCNet_74663711474467
// MI455X (gfx1250) — matrix operand order repaired and checked by reading, not compiled, not run on hardware
//
#include <hip/hip_runtime.h>
#include <stdint.h>

// ---------------------------------------------------------------------------
// CDNA5 (gfx1250) PWC-Net: convs dominate (compute-bound, AI ~1e3 FLOP/B) ->
// implicit GEMM on v_wmma_f32_16x16x32_f16, f16 activations (fits in 192MB L2).
// B (weight) tiles move via the Tensor Data Mover (tensor_load_to_lds with a
// hand-built D#: 2D tile, LDS pad -> 80B pitch, HW zero-fill for OOB rows),
// bias is staged with global_load_async_to_lds_b128 overlapped with the whole
// K loop. Fallbacks keep the kernel compiling on any toolchain.
// ---------------------------------------------------------------------------

typedef _Float16 v16h __attribute__((ext_vector_type(16)));
typedef _Float16 v8h  __attribute__((ext_vector_type(8)));
typedef float    v8f  __attribute__((ext_vector_type(8)));
typedef int      v4i  __attribute__((ext_vector_type(4)));
typedef int      v8i  __attribute__((ext_vector_type(8)));
typedef unsigned v4u  __attribute__((ext_vector_type(4)));

#if defined(__has_builtin)
#  if __has_builtin(__builtin_amdgcn_global_load_async_to_lds_b128)
#    define HAVE_ASYNC_LDS 1
#  endif
#  if __has_builtin(__builtin_amdgcn_tensor_load_to_lds) && \
      __has_builtin(__builtin_amdgcn_s_wait_tensorcnt)
#    define HAVE_TDM 1
#  endif
#endif
#ifndef HAVE_ASYNC_LDS
#  define HAVE_ASYNC_LDS 0
#endif
#ifndef HAVE_TDM
#  define HAVE_TDM 0
#endif

#if HAVE_ASYNC_LDS
typedef __attribute__((address_space(1))) v4i ga_v4i;   // global int4
typedef __attribute__((address_space(3))) v4i lds_v4i;  // LDS int4
#endif

__device__ __forceinline__ void wait_async_lds() {
#if HAVE_ASYNC_LDS
#  if defined(__has_builtin) && __has_builtin(__builtin_amdgcn_s_wait_asynccnt)
  __builtin_amdgcn_s_wait_asynccnt(0);
#  else
  asm volatile("s_wait_asynccnt 0" ::: "memory");
#  endif
#endif
}

#if HAVE_TDM
__device__ __forceinline__ unsigned lds_off_u32(void* p) {
  return (unsigned)(size_t)(__attribute__((address_space(3))) void*)p;
}
#endif

// ---------------------------------------------------------------------------
// f32 -> f16 convert with K zero-padding: dst [rows][Kpad], src [rows][Ktot].
// ---------------------------------------------------------------------------
__global__ void cvt_f16_pad(const float* __restrict__ src, _Float16* __restrict__ dst,
                            int rows, int Ktot, int Kpad) {
  int i = blockIdx.x * 256 + threadIdx.x;
  int total = rows * Kpad;
  if (i >= total) return;
  int r = i / Kpad, k = i % Kpad;
  dst[i] = (k < Ktot) ? (_Float16)src[r * Ktot + k] : (_Float16)0.f;
}

// ---------------------------------------------------------------------------
// Implicit-GEMM conv (k x k, stride, dilation, leaky-ReLU) via WMMA f16.
//   M = B*OH*OW, N = Cout, K = Cin*ks*ks (zero-padded to Kpad).
// Block = 256 thr = 8 waves; tile 64(M) x 64(N); wave = 16(M) x 32(N).
// ---------------------------------------------------------------------------
#define APITCH 40   // halves per LDS row (80B: b128-aligned, banks spread)

__global__ __launch_bounds__(256)
void conv_wmma(const _Float16* __restrict__ act, const _Float16* __restrict__ wgt,
               const float* __restrict__ bias,
               float* __restrict__ outF, _Float16* __restrict__ outH,
               int Bn, int Cin, int IH, int IW, int Cout, int OH, int OW,
               int stride, int dil, int relu, int ks, int Kpad) {
  __shared__ _Float16 Alds[64 * APITCH];
  __shared__ _Float16 Blds[64 * APITCH];
  __shared__ __align__(16) float biasLds[64];

  const int t = threadIdx.x;
  const int lane = t & 31;
  const int wave = t >> 5;
  const int wm = wave & 3;          // 4 M sub-tiles of 16
  const int wn = wave >> 2;         // 2 N groups of 32
  const int taps = ks * ks;
  const int Npix = Bn * OH * OW;
  const int mBase = blockIdx.x * 64;
  const int nBase = blockIdx.y * 64;
  const int hw = OH * OW;
  const int kc = (ks - 1) >> 1;
  const int chw = IH * IW;

  // ---- bias -> LDS; full 16B chunks via async-to-LDS (wave 0), rest scalar --
  {
    int nfull = 0;
    if (bias) { int rem = Cout - nBase; if (rem > 64) rem = 64; if (rem > 0) nfull = rem & ~3; }
#if HAVE_ASYNC_LDS
    if (wave == 0 && lane < (nfull >> 2)) {
      const float* bsrc = bias + nBase + lane * 4;
      __builtin_amdgcn_global_load_async_to_lds_b128(
          (ga_v4i*)bsrc, (lds_v4i*)&biasLds[lane * 4], 0, 0);
    }
    if (t >= nfull && t < 64) {
      int n = nBase + t;
      biasLds[t] = (bias && n < Cout) ? bias[n] : 0.f;
    }
#else
    (void)nfull;
    if (t < 64) {
      int n = nBase + t;
      biasLds[t] = (bias && n < Cout) ? bias[n] : 0.f;
    }
#endif
  }

  // ---- per-thread im2col invariants: thread owns pixel m = t&63, 8 K slots --
  const int m  = t & 63;
  const int t6 = t >> 6;                     // 0..3 : kl = i*4 + t6
  const int mg = mBase + m;
  const bool validM = (mg < Npix);
  int bI = 0, rowr = 0;
  if (validM) { bI = mg / hw; rowr = mg - bI * hw; }
  const int oyS = (rowr / OW) * stride;
  const int oxS = (rowr - (rowr / OW) * OW) * stride;
  const _Float16* actB = act + (size_t)bI * Cin * chw;
  // mixed-radix counter over (cin, ky, kx); step = +4 in kx digit per slot
  int cin = t6 / taps;
  int rem = t6 - cin * taps;
  int ky = rem / ks, kx = rem - (rem / ks) * ks;
  const bool kIs1 = (taps == 1);

  v8f acc0 = {}, acc1 = {};

  for (int kb = 0; kb < Kpad; kb += 32) {
    // ---- issue TDM for the 64x32 B tile first (overlaps the A gather) ----
#if HAVE_TDM
    if (wave == 0) {
      uint64_t ga = (uint64_t)(uintptr_t)(wgt + (size_t)nBase * Kpad + kb);
      int rows = Cout - nBase; if (rows < 0) rows = 0;      // HW zero-fills OOB
      int cols = Kpad - kb;                                  // >= 32 always
      v4u g0 = { 1u,                                         // count=1
                 lds_off_u32((void*)Blds),                   // lds_addr (bytes)
                 (unsigned)ga,
                 (unsigned)((ga >> 32) & 0x01FFFFFFu) | (2u << 30) };  // type=2
      v8i g1 = { (int)((1u << 16) |                          // data_size=2B
                       (1u << 20) |                          // pad_enable
                       (3u << 22) |                          // pad_interval: 16 DW
                       (3u << 25)),                          // pad_amount:   4 DW
                 (int)((cols & 0xFFFF) << 16),               // tensor_dim0 lo
                 (int)(((cols >> 16) & 0xFFFF) | ((rows & 0xFFFF) << 16)),
                 (int)(((rows >> 16) & 0xFFFF) | (32u << 16)),  // tile_dim0=32
                 64,                                         // tile_dim1=64
                 Kpad,                                       // dim0 stride lo
                 0, 0 };
      v4i g2 = { 0, 0, 0, 0 };
      v4i g3 = { 0, 0, 0, 0 };
      v8i g4 = { 0, 0, 0, 0, 0, 0, 0, 0 };
      __builtin_amdgcn_tensor_load_to_lds(g0, g1, g2, g3, g4, 0);
    }
#endif
    // ---- stage A (im2col gather): slot i covers kl = i*4 + t6 ----
#pragma unroll
    for (int i = 0; i < 8; ++i) {
      const int kl = (i << 2) + t6;
      const int iy = oyS + (ky - kc) * dil;
      const int ix = oxS + (kx - kc) * dil;
      _Float16 hv = (_Float16)0.f;
      if (validM && (unsigned)cin < (unsigned)Cin &&
          (unsigned)iy < (unsigned)IH && (unsigned)ix < (unsigned)IW)
        hv = actB[(size_t)cin * chw + iy * IW + ix];
      Alds[m * APITCH + kl] = hv;
      if (kIs1) {
        cin += 4;
      } else {
        kx += 4;
        if (kx >= ks) { kx -= ks; ++ky; }
        if (kx >= ks) { kx -= ks; ++ky; }
        if (kx >= ks) { kx -= ks; ++ky; }
        if (ky >= ks) { ky -= ks; ++cin; }
        if (ky >= ks) { ky -= ks; ++cin; }
      }
    }
    // ---- finish B staging ----
#if HAVE_TDM
    if (wave == 0) __builtin_amdgcn_s_wait_tensorcnt(0);
#else
    {
      int n = t >> 2, ch = t & 3;   // 64 rows x 4 x 16B chunks
      const _Float16* src = wgt + (size_t)(nBase + n) * Kpad + kb + ch * 8;
      _Float16* dst = &Blds[n * APITCH + ch * 8];
      bool inb = (nBase + n) < Cout;
#  if HAVE_ASYNC_LDS
      if (inb) {
        __builtin_amdgcn_global_load_async_to_lds_b128((ga_v4i*)src, (lds_v4i*)dst, 0, 0);
      } else {
        v8h z = {};
        *(v8h*)dst = z;
      }
#  else
      v8h val = {};
      if (inb) val = *(const v8h*)src;
      *(v8h*)dst = val;
#  endif
      if (kb + 32 < Kpad) __builtin_prefetch(src + 32, 0, 0);  // global_prefetch_b8
      wait_async_lds();
    }
#endif
    __syncthreads();

    // ---- WMMA: per-lane loads, same contiguous k = 16*(lane/16) + s in A and B ----
    {
      const int row = wm * 16 + (lane & 15);
      const int k0  = (lane < 16) ? 0 : 16;              // A: chunks at k0, k0+8
      const _Float16* ap = &Alds[row * APITCH + k0];
      v8h a0 = *(const v8h*)(ap);
      v8h a1 = *(const v8h*)(ap + 8);
      v16h av = __builtin_shufflevector(a0, a1, 0, 1, 2, 3, 4, 5, 6, 7,
                                        8, 9, 10, 11, 12, 13, 14, 15);
      const int kob = (lane < 16) ? 0 : 16;              // B: one 16-half run
      v16h b0 = *(const v16h*)(&Blds[(wn * 32 + (lane & 15)) * APITCH + kob]);
      v16h b1 = *(const v16h*)(&Blds[(wn * 32 + 16 + (lane & 15)) * APITCH + kob]);
      acc0 = __builtin_amdgcn_wmma_f32_16x16x32_f16(false, av, false, b0,
                                                    (short)0, acc0, false, false);
      acc1 = __builtin_amdgcn_wmma_f32_16x16x32_f16(false, av, false, b1,
                                                    (short)0, acc1, false, false);
    }
    __syncthreads();
  }

  wait_async_lds();     // bias async (wave 0); no-op elsewhere
  __syncthreads();

  // ---- epilogue: bias + leaky ReLU; incremental pixel decomposition ----
  {
    v8f accs[2] = {acc0, acc1};
    const int n0 = wn * 32 + (lane & 15);
    const float bv[2] = { biasLds[n0], biasLds[n0 + 16] };
    int m0 = mBase + wm * 16 + ((lane & 16) ? 8 : 0);
    int bb_i = m0 / hw;
    int rowoff = m0 - bb_i * hw;
    size_t bb = (size_t)bb_i * Cout * hw;
#pragma unroll
    for (int r = 0; r < 8; ++r) {
      if (m0 + r < Npix) {
#pragma unroll
        for (int p = 0; p < 2; ++p) {
          int n = nBase + n0 + p * 16;
          if (n < Cout) {
            float v = accs[p][r] + bv[p];
            if (relu) v = (v > 0.f) ? v : 0.1f * v;
            size_t o = bb + (size_t)n * hw + rowoff;
            if (outF) outF[o] = v;
            if (outH) outH[o] = (_Float16)v;
          }
        }
      }
      if (++rowoff == hw) { rowoff = 0; bb += (size_t)Cout * hw; }
    }
  }
}

// ---------------------------------------------------------------------------
// 81-shift correlation cost volume with leaky ReLU (bandwidth-bound VALU).
// ---------------------------------------------------------------------------
__global__ void cost_volume_leaky(const _Float16* __restrict__ f1,
                                  const _Float16* __restrict__ f2,
                                  float* __restrict__ out,
                                  int Bn, int C, int H, int W, int md) {
  int D = 2 * md + 1;
  int i = blockIdx.x * 256 + threadIdx.x;
  int tot = Bn * D * D * H * W;
  if (i >= tot) return;
  int x = i % W, y = (i / W) % H;
  int s = (i / (W * H)) % (D * D);
  int b = i / (W * H * D * D);
  int dy = s / D - md, dx = s % D - md;
  int yy = y + dy, xx = x + dx;
  float acc = 0.f;
  if ((unsigned)yy < (unsigned)H && (unsigned)xx < (unsigned)W) {
    const _Float16* p1 = f1 + ((size_t)b * C * H + y) * W + x;
    const _Float16* p2 = f2 + ((size_t)b * C * H + yy) * W + xx;
    __builtin_prefetch(p2, 0, 0);
    for (int c = 0; c < C; ++c)
      acc += (float)p1[(size_t)c * H * W] * (float)p2[(size_t)c * H * W];
    acc /= (float)C;
  }
  out[i] = (acc > 0.f) ? acc : 0.1f * acc;
}

// ---------------------------------------------------------------------------
// Align-corners bilinear resize of the first Cuse channels of [B,Cin,h1,w1].
// ---------------------------------------------------------------------------
__global__ void resize_subch(const float* __restrict__ in, float* __restrict__ out,
                             int Bn, int Cin, int Cuse, int h1, int w1, int h2, int w2) {
  int i = blockIdx.x * 256 + threadIdx.x;
  int tot = Bn * Cuse * h2 * w2;
  if (i >= tot) return;
  int x = i % w2, y = (i / w2) % h2;
  int c = (i / (w2 * h2)) % Cuse, b = i / (w2 * h2 * Cuse);
  float fy = (h2 > 1) ? (float)y * (float)(h1 - 1) / (float)(h2 - 1) : 0.f;
  float fx = (w2 > 1) ? (float)x * (float)(w1 - 1) / (float)(w2 - 1) : 0.f;
  int y0 = (int)fy, x0 = (int)fx;
  int y1 = (y0 + 1 < h1) ? y0 + 1 : h1 - 1;
  int x1 = (x0 + 1 < w1) ? x0 + 1 : w1 - 1;
  float wy = fy - (float)y0, wx = fx - (float)x0;
  const float* p = in + ((size_t)(b * Cin + c) * h1) * w1;
  out[i] = p[y0 * w1 + x0] * (1.f - wy) * (1.f - wx) +
           p[y0 * w1 + x1] * (1.f - wy) * wx +
           p[y1 * w1 + x0] * wy * (1.f - wx) +
           p[y1 * w1 + x1] * wy * wx;
}

__global__ void fill_zero(float* p, int n) {
  int i = blockIdx.x * 256 + threadIdx.x;
  if (i < n) p[i] = 0.f;
}

// ---------------------------------------------------------------------------
// Host orchestration (all on `stream`, workspace bump-allocated from d_ws).
// ---------------------------------------------------------------------------
static inline size_t rnd256(size_t x) { return (x + 255) & ~(size_t)255; }

extern "C" void kernel_launch(void* const* d_in, const int* in_sizes, int n_in,
                              void* d_out, int out_size, void* d_ws, size_t ws_size,
                              hipStream_t stream) {
  (void)in_sizes;
  const int Bn = 2, H = 192, W = 448;
  int chs[7] = {3, 16, 32, 64, 96, 128, 196};
  int hs[7], wz[7];
  hs[0] = H; wz[0] = W;
  for (int l = 0; l < 6; ++l) { hs[l + 1] = hs[l] / 2; wz[l + 1] = wz[l] / 2; }

  // --- workspace layout ---
  size_t off = 0;
  auto alloc = [&](size_t bytes) { size_t o = off; off += rnd256(bytes); return o; };
  size_t o_img[2];
  o_img[0] = alloc((size_t)Bn * 3 * H * W * 2);
  o_img[1] = alloc((size_t)Bn * 3 * H * W * 2);
  size_t o_w[6][2];
  int kpad[6][2];
  for (int l = 0; l < 6; ++l)
    for (int j = 0; j < 2; ++j) {
      int Cin = j ? chs[l + 1] : chs[l];
      int Kt = Cin * 9, Kp = ((Kt + 31) / 32) * 32;
      kpad[l][j] = Kp;
      o_w[l][j] = alloc((size_t)chs[l + 1] * Kp * 2);
    }
  size_t o_tmp = alloc((size_t)Bn * 16 * hs[1] * wz[1] * 2);  // largest intermediate
  size_t o_lvl[2][6];
  for (int img = 0; img < 2; ++img)
    for (int l = 0; l < 6; ++l)
      o_lvl[img][l] = alloc((size_t)Bn * chs[l + 1] * hs[l + 1] * wz[l + 1] * 2);
  size_t corrN = (size_t)Bn * 81 * hs[6] * wz[6];
  size_t o_corr = alloc(corrN * 4);

  fill_zero<<<(out_size + 255) / 256, 256, 0, stream>>>((float*)d_out, out_size);
  if (off > ws_size || n_in < 26) return;
  char* wsb = (char*)d_ws;

  // --- convert inputs + feature-extractor weights to f16 (K zero-padded) ---
  for (int img = 0; img < 2; ++img) {
    int n = Bn * 3 * H * W;
    cvt_f16_pad<<<(n + 255) / 256, 256, 0, stream>>>(
        (const float*)d_in[img], (_Float16*)(wsb + o_img[img]), 1, n, n);
  }
  for (int l = 0; l < 6; ++l)
    for (int j = 0; j < 2; ++j) {
      int Cout = chs[l + 1];
      int Kt = (j ? chs[l + 1] : chs[l]) * 9, Kp = kpad[l][j];
      int n = Cout * Kp;
      cvt_f16_pad<<<(n + 255) / 256, 256, 0, stream>>>(
          (const float*)d_in[2 + l * 4 + j * 2], (_Float16*)(wsb + o_w[l][j]), Cout, Kt, Kp);
    }

  // --- both feature pyramids: 12 WMMA convs each ---
  for (int img = 0; img < 2; ++img) {
    const _Float16* cur = (const _Float16*)(wsb + o_img[img]);
    int ci = 3, ih = H, iw = W;
    for (int l = 0; l < 6; ++l) {
      int co = chs[l + 1], oh = hs[l + 1], ow = wz[l + 1];
      int npix = Bn * oh * ow;
      dim3 g((npix + 63) / 64, (co + 63) / 64);
      conv_wmma<<<g, 256, 0, stream>>>(
          cur, (const _Float16*)(wsb + o_w[l][0]), (const float*)d_in[2 + l * 4 + 1],
          nullptr, (_Float16*)(wsb + o_tmp),
          Bn, ci, ih, iw, co, oh, ow, /*stride=*/2, /*dil=*/1, /*relu=*/1, 3, kpad[l][0]);
      conv_wmma<<<g, 256, 0, stream>>>(
          (const _Float16*)(wsb + o_tmp), (const _Float16*)(wsb + o_w[l][1]),
          (const float*)d_in[2 + l * 4 + 3],
          nullptr, (_Float16*)(wsb + o_lvl[img][l]),
          Bn, co, oh, ow, co, oh, ow, /*stride=*/1, /*dil=*/1, /*relu=*/1, 3, kpad[l][1]);
      cur = (const _Float16*)(wsb + o_lvl[img][l]);
      ci = co; ih = oh; iw = ow;
    }
  }

  // --- coarsest-level 81-shift correlation + leaky ---
  {
    int Hc = hs[6], Wc = wz[6];
    int tot = (int)corrN;
    cost_volume_leaky<<<(tot + 255) / 256, 256, 0, stream>>>(
        (const _Float16*)(wsb + o_lvl[0][5]), (const _Float16*)(wsb + o_lvl[1][5]),
        (float*)(wsb + o_corr), Bn, 196, Hc, Wc, 4);
  }

  // --- upsample 3 channels into d_out [B,3,H,W] ---
  {
    int tot = Bn * 3 * H * W;
    if (tot > out_size) tot = out_size;
    resize_subch<<<(tot + 255) / 256, 256, 0, stream>>>(
        (const float*)(wsb + o_corr), (float*)d_out, Bn, 81, 3, hs[6], wz[6], H, W);
  }
}
